// QFCN_20057497272455
// MI455X (gfx1250) — matrix operand order repaired and checked by reading, not compiled, not run on hardware
//
#include <hip/hip_runtime.h>

typedef __attribute__((ext_vector_type(16))) _Float16 v16h;
typedef __attribute__((ext_vector_type(8)))  float    v8f;

// ---------------------------------------------------------------------------
// Prep kernel: build U1 (16x16 complex) and U2 (4x4 complex) from the small
// parameter vectors.  One wave; thread t<16 simulates column t of U1 as a
// statevector, threads 16..19 simulate columns of U2.
// ws layout (floats): [0,256) U1r row-major, [256,512) U1i,
//                     [512,528) U2r row-major, [528,544) U2i.
// ---------------------------------------------------------------------------
__global__ void qfcn_prep(const float* __restrict__ f1, const float* __restrict__ p1,
                          const float* __restrict__ f2, const float* __restrict__ p2,
                          float* __restrict__ ws) {
    const int t = threadIdx.x;
    if (t < 16) {
        float vr[16], vi[16], wr[16], wi[16];
        for (int j = 0; j < 16; ++j) { vr[j] = (j == t) ? 1.f : 0.f; vi[j] = 0.f; }

        const float STEP = 0.39269908169872414f; // 2*pi/16
        auto qft16 = [&](float sgn) {
            for (int j = 0; j < 16; ++j) {
                float ar = 0.f, ai = 0.f;
                for (int k = 0; k < 16; ++k) {
                    float s, c;
                    __sincosf(sgn * STEP * (float)((j * k) & 15), &s, &c);
                    ar += c * vr[k] - s * vi[k];
                    ai += c * vi[k] + s * vr[k];
                }
                wr[j] = 0.25f * ar; wi[j] = 0.25f * ai;
            }
            for (int j = 0; j < 16; ++j) { vr[j] = wr[j]; vi[j] = wi[j]; }
        };
        auto phase = [&](int j, float ph) {
            float s, c; __sincosf(ph, &s, &c);
            float nr = vr[j] * c - vi[j] * s;
            float ni = vi[j] * c + vr[j] * s;
            vr[j] = nr; vi[j] = ni;
        };
        auto crz = [&](int cmask, int tmask, float th) {
            for (int j = 0; j < 16; ++j)
                if (j & cmask) phase(j, ((j & tmask) ? 0.5f : -0.5f) * th);
        };
        auto xg = [&](int mask) {
            for (int j = 0; j < 16; ++j)
                if (!(j & mask)) {
                    int j2 = j | mask;
                    float a = vr[j]; vr[j] = vr[j2]; vr[j2] = a;
                    float b = vi[j]; vi[j] = vi[j2]; vi[j2] = b;
                }
        };
        auto crx = [&](int cmask, int tmask, float th) {
            float s, c; __sincosf(0.5f * th, &s, &c);
            for (int j = 0; j < 16; ++j)
                if ((j & cmask) && !(j & tmask)) {
                    int j2 = j | tmask;
                    float r0 = vr[j], i0 = vi[j], r1 = vr[j2], i1 = vi[j2];
                    vr[j]  = c * r0 + s * i1;  vi[j]  = c * i0 - s * r1;
                    vr[j2] = c * r1 + s * i0;  vi[j2] = c * i1 - s * r0;
                }
        };

        qft16(1.f);
        for (int i = 0; i < 4; ++i) {
            float th = f1[i];
            for (int j = 0; j < 16; ++j)
                phase(j, (((j >> (3 - i)) & 1) ? 0.5f : -0.5f) * th);
        }
        qft16(-1.f);
        crz(8, 4, p1[0]); xg(8); crx(8, 4, p1[1]); xg(8);   // wires 0(ctrl)/1(tgt)
        crz(2, 1, p1[0]); xg(2); crx(2, 1, p1[1]); xg(2);   // wires 2(ctrl)/3(tgt)

        for (int j = 0; j < 16; ++j) {
            ws[j * 16 + t]       = vr[j];
            ws[256 + j * 16 + t] = vi[j];
        }
    } else if (t < 20) {
        const int cc = t - 16;
        float vr[4], vi[4], wr[4], wi[4];
        for (int j = 0; j < 4; ++j) { vr[j] = (j == cc) ? 1.f : 0.f; vi[j] = 0.f; }

        const float STEP4 = 1.5707963267948966f; // 2*pi/4
        auto qft4 = [&](float sgn) {
            for (int j = 0; j < 4; ++j) {
                float ar = 0.f, ai = 0.f;
                for (int k = 0; k < 4; ++k) {
                    float s, c;
                    __sincosf(sgn * STEP4 * (float)((j * k) & 3), &s, &c);
                    ar += c * vr[k] - s * vi[k];
                    ai += c * vi[k] + s * vr[k];
                }
                wr[j] = 0.5f * ar; wi[j] = 0.5f * ai;
            }
            for (int j = 0; j < 4; ++j) { vr[j] = wr[j]; vi[j] = wi[j]; }
        };
        auto phase = [&](int j, float ph) {
            float s, c; __sincosf(ph, &s, &c);
            float nr = vr[j] * c - vi[j] * s;
            float ni = vi[j] * c + vr[j] * s;
            vr[j] = nr; vi[j] = ni;
        };

        qft4(1.f);
        for (int j = 0; j < 4; ++j) phase(j, (((j >> 1) & 1) ? 0.5f : -0.5f) * f2[0]);
        for (int j = 0; j < 4; ++j) phase(j, ((j & 1) ? 0.5f : -0.5f) * f2[1]);
        qft4(-1.f);
        for (int j = 0; j < 4; ++j)                     // ctrl(rz(p2[0])) wire0
            if (j & 2) phase(j, ((j & 1) ? 0.5f : -0.5f) * p2[0]);
        { float a, b;                                   // X wire0
          a = vr[0]; vr[0] = vr[2]; vr[2] = a; b = vi[0]; vi[0] = vi[2]; vi[2] = b;
          a = vr[1]; vr[1] = vr[3]; vr[3] = a; b = vi[1]; vi[1] = vi[3]; vi[3] = b; }
        { float s, c; __sincosf(0.5f * p2[1], &s, &c);  // ctrl(rx(p2[1])) wire0
          float r0 = vr[2], i0 = vi[2], r1 = vr[3], i1 = vi[3];
          vr[2] = c * r0 + s * i1;  vi[2] = c * i0 - s * r1;
          vr[3] = c * r1 + s * i0;  vi[3] = c * i1 - s * r0; }
        { float a, b;                                   // X wire0
          a = vr[0]; vr[0] = vr[2]; vr[2] = a; b = vi[0]; vi[0] = vi[2]; vi[2] = b;
          a = vr[1]; vr[1] = vr[3]; vr[3] = a; b = vi[1]; vi[1] = vi[3]; vi[3] = b; }

        for (int j = 0; j < 4; ++j) {
            ws[512 + j * 4 + cc] = vr[j];
            ws[528 + j * 4 + cc] = vi[j];
        }
    }
}

// ---------------------------------------------------------------------------
// Main kernel: one wave handles 16 samples per tile, transposed orientation:
//   D[16 states, 16 samples]_{r,i} = U1{r,i} @ psi^T
// A = U1 (constant fragments, loaded once), B = psi^T (lanes 0-15 carry one
// full sample each; lanes 16-31 are the K=16..31 zero padding).
// C/D layout puts the sample on the lane and the state on the register, so
// the +-1-weighted |psi|^2 reductions are in-register with compile-time signs
// plus a single shfl_xor(16) to merge the two state halves.
// ---------------------------------------------------------------------------
__global__ __launch_bounds__(256) void qfcn_main(const float* __restrict__ x,
                                                 const float* __restrict__ Wm,
                                                 const float* __restrict__ bv,
                                                 const float* __restrict__ ws,
                                                 float* __restrict__ out,
                                                 int numTiles) {
    const int lane = threadIdx.x & 31;
    const int waveInBlock = threadIdx.x >> 5;
    const int wavesPerBlock = blockDim.x >> 5;
    const int globalWave = blockIdx.x * wavesPerBlock + waveInBlock;
    const int totalWaves = gridDim.x * wavesPerBlock;

    const int n   = lane & 15;   // A: state row M.  B/C: sample column N.
    const int grp = lane >> 4;   // A: K-half.       B: grp1 = zero padding.

    // --- A fragments: U1r/U1i rows as f16 (constant per wave) ---
    // lane L: M = L&15, grp 0: elements 0..15 = K = 0..15; grp 1 (K=16..31) = 0.
    v16h ar = {}; v16h ai = {};
#pragma unroll
    for (int j = 0; j < 16; ++j) {
        ar[j] = grp ? (_Float16)0.f : (_Float16)ws[n * 16 + j];
        ai[j] = grp ? (_Float16)0.f : (_Float16)ws[256 + n * 16 + j];
    }

    // --- uniform layer-2 constants (scalar loads) ---
    float u2r[16], u2i[16];
#pragma unroll
    for (int i = 0; i < 16; ++i) { u2r[i] = ws[512 + i]; u2i[i] = ws[528 + i]; }
    const float W0 = Wm[0], W1 = Wm[1], B0 = bv[0], B1 = bv[1];

    const float bmask = (grp == 0) ? 1.0f : 0.0f;  // zero out K=16..31 B lanes

    for (int tile = globalWave; tile < numTiles; tile += totalWaves) {
        // Lane n holds sample (tile*16+n); upper half duplicates the same
        // addresses (coalesces onto the same cachelines) and is masked to 0.
        const float* xs = x + ((size_t)(tile * 16 + n)) * 16;
        float4 q0 = *(const float4*)(xs);
        float4 q1 = *(const float4*)(xs + 4);
        float4 q2 = *(const float4*)(xs + 8);
        float4 q3 = *(const float4*)(xs + 12);

        float ssq = q0.x * q0.x + q0.y * q0.y + q0.z * q0.z + q0.w * q0.w
                  + q1.x * q1.x + q1.y * q1.y + q1.z * q1.z + q1.w * q1.w
                  + q2.x * q2.x + q2.y * q2.y + q2.z * q2.z + q2.w * q2.w
                  + q3.x * q3.x + q3.y * q3.y + q3.z * q3.z + q3.w * q3.w;
        const float sc = rsqrtf(ssq) * bmask;

        v16h b;
        b[0]  = (_Float16)(q0.x * sc); b[1]  = (_Float16)(q0.y * sc);
        b[2]  = (_Float16)(q0.z * sc); b[3]  = (_Float16)(q0.w * sc);
        b[4]  = (_Float16)(q1.x * sc); b[5]  = (_Float16)(q1.y * sc);
        b[6]  = (_Float16)(q1.z * sc); b[7]  = (_Float16)(q1.w * sc);
        b[8]  = (_Float16)(q2.x * sc); b[9]  = (_Float16)(q2.y * sc);
        b[10] = (_Float16)(q2.z * sc); b[11] = (_Float16)(q2.w * sc);
        b[12] = (_Float16)(q3.x * sc); b[13] = (_Float16)(q3.y * sc);
        b[14] = (_Float16)(q3.z * sc); b[15] = (_Float16)(q3.w * sc);

        v8f cr = {}; v8f ci = {};
        cr = __builtin_amdgcn_wmma_f32_16x16x32_f16(false, ar, false, b,
                                                    (short)0, cr, false, false);
        ci = __builtin_amdgcn_wmma_f32_16x16x32_f16(false, ai, false, b,
                                                    (short)0, ci, false, false);

        // cr[v],ci[v] = state m = v + 8*grp of sample n.  Signs are
        // compile-time: s1(m) = -(bit2 of v), s3(m) = -(bit0 of v); the
        // 8*grp offset only touches bit3, which neither sign uses.
        float e1p = 0.f, e3p = 0.f;
#pragma unroll
        for (int v = 0; v < 8; ++v) {
            float p = cr[v] * cr[v] + ci[v] * ci[v];
            e1p += (v & 4) ? -p : p;
            e3p += (v & 1) ? -p : p;
        }
        const float e1 = e1p + __shfl_xor(e1p, 16, 32);  // merge state halves
        const float e3 = e3p + __shfl_xor(e3p, 16, 32);

        if (grp == 0) {   // each low-half lane owns sample tile*16 + n
            float c0, sn0, c1, sn1;
            __sincosf(0.5f * e1, &sn0, &c0);
            __sincosf(0.5f * e3, &sn1, &c1);
            float p2v[4] = { c0 * c1, c0 * sn1, sn0 * c1, sn0 * sn1 };
            float e = 0.f;
#pragma unroll
            for (int i = 0; i < 4; ++i) {
                float orr = 0.f, oii = 0.f;
#pragma unroll
                for (int j = 0; j < 4; ++j) {
                    orr += u2r[i * 4 + j] * p2v[j];
                    oii += u2i[i * 4 + j] * p2v[j];
                }
                float pp = orr * orr + oii * oii;
                e += (i & 1) ? -pp : pp;            // z_signs(wire1, n=2)
            }
            float2 res;
            res.x = e * W0 + B0;
            res.y = e * W1 + B1;
            *(float2*)(out + ((size_t)(tile * 16 + n)) * 2) = res;
        }
    }
}

extern "C" void kernel_launch(void* const* d_in, const int* in_sizes, int n_in,
                              void* d_out, int out_size, void* d_ws, size_t ws_size,
                              hipStream_t stream) {
    const float* x  = (const float*)d_in[0];
    const float* f1 = (const float*)d_in[1];
    const float* p1 = (const float*)d_in[2];
    const float* f2 = (const float*)d_in[3];
    const float* p2 = (const float*)d_in[4];
    const float* W  = (const float*)d_in[5];
    const float* b  = (const float*)d_in[6];
    float* out = (float*)d_out;
    float* ws  = (float*)d_ws;   // 544 floats used

    const int B = in_sizes[0] / 16;      // samples
    const int numTiles = B / 16;         // 16 samples per wave-tile

    qfcn_prep<<<1, 32, 0, stream>>>(f1, p1, f2, p2, ws);

    int blocks = (numTiles + 7) / 8;     // 8 waves per block
    if (blocks > 2048) blocks = 2048;
    if (blocks < 1) blocks = 1;
    qfcn_main<<<blocks, 256, 0, stream>>>(x, W, b, ws, out, numTiles);
}
